// MoEMLP_34995393528501
// MI455X (gfx1250) — matrix operand order repaired and checked by reading, not compiled, not run on hardware
//
#include <hip/hip_runtime.h>

// MoE MLP for MI455X (gfx1250, wave32).
// B=4, T=2048 -> N=8192 tokens; D=1024; H=4096; E=8; TOP_K=2.
//
// Pipeline:
//   K0 moe_zero   : zero d_out (atomic-combined) + expert counters
//   K1 moe_gate   : per-token logits, top-2 softmax, compact per-expert token
//                   lists + gate weights into d_ws
//   K2 moe_expert : per (expert, 32-token tile) fused fc+act+proj using
//                   v_wmma_f32_16x16x32_bf16; f32 loaded as b128, converted
//                   with v_cvt_pk_bf16_f32, staged to LDS as b128; gate folded
//                   into activations; f32 global atomic combine.
//
// Workspace layout (needs ~513 KB):
//   [0,32)                 int   counts[8]
//   [256, 256+256K)        int   lists[E][N]
//   [256+256K, 256+512K)   float gatew[E][N]

#define NTOK 8192
#define DDIM 1024
#define HDIM 4096
#define NEXP 8
#define TM   32     // tokens per expert tile

typedef __attribute__((ext_vector_type(16))) __bf16       v16bf;
typedef __attribute__((ext_vector_type(8)))  float        v8f;
typedef __attribute__((ext_vector_type(4)))  float        f32x4;
typedef __attribute__((ext_vector_type(4)))  unsigned int u32x4;

union Frag { v16bf v; u32x4 q[2]; };

// f32 pair -> packed bf16x2 via the hardware packed converter (RNE)
__device__ __forceinline__ unsigned pack2(float lo, float hi) {
  unsigned r;
  asm("v_cvt_pk_bf16_f32 %0, %1, %2" : "=v"(r) : "v"(lo), "v"(hi));
  return r;
}

// load 8 consecutive f32 (two b128 loads) -> 8 packed bf16 in one u32x4
__device__ __forceinline__ u32x4 pack8(const float* __restrict__ g) {
  f32x4 v0 = *(const f32x4*)g;
  f32x4 v1 = *(const f32x4*)(g + 4);
  u32x4 r;
  r.x = pack2(v0.x, v0.y);
  r.y = pack2(v0.z, v0.w);
  r.z = pack2(v1.x, v1.y);
  r.w = pack2(v1.z, v1.w);
  return r;
}

// ---------------------------------------------------------------- K0: zero
__global__ void moe_zero(float* __restrict__ out, int n, int* __restrict__ counts) {
  int idx = blockIdx.x * blockDim.x + threadIdx.x;
  for (int i = idx; i < n; i += gridDim.x * blockDim.x) out[i] = 0.f;
  if (idx < NEXP) counts[idx] = 0;
}

// ---------------------------------------------------------------- K1: gating
__global__ void __launch_bounds__(256)
moe_gate(const float* __restrict__ x, const float* __restrict__ Wg,
         int* __restrict__ counts, int* __restrict__ lists,
         float* __restrict__ gatew) {
  int wave = threadIdx.x >> 5;
  int lane = threadIdx.x & 31;
  int tok  = blockIdx.x * 8 + wave;
  if (tok >= NTOK) return;
  const float* xr = x + (size_t)tok * DDIM;

  float s[NEXP];
#pragma unroll
  for (int e = 0; e < NEXP; ++e) s[e] = 0.f;
#pragma unroll
  for (int i = 0; i < DDIM / 128; ++i) {          // lane-parallel b128 dots
    int d = (lane + i * 32) * 4;
    f32x4 xv = *(const f32x4*)(xr + d);
#pragma unroll
    for (int e = 0; e < NEXP; ++e) {
      f32x4 wv = *(const f32x4*)(Wg + e * DDIM + d);
      s[e] += xv.x * wv.x + xv.y * wv.y + xv.z * wv.z + xv.w * wv.w;
    }
  }
#pragma unroll
  for (int e = 0; e < NEXP; ++e) {
#pragma unroll
    for (int off = 16; off > 0; off >>= 1) s[e] += __shfl_xor(s[e], off, 32);
  }
  if (lane == 0) {
    int e0 = 0; float l0 = s[0];
#pragma unroll
    for (int e = 1; e < NEXP; ++e) if (s[e] > l0) { l0 = s[e]; e0 = e; }
    int e1 = -1; float l1 = -3.4e38f;
#pragma unroll
    for (int e = 0; e < NEXP; ++e) if (e != e0 && s[e] > l1) { l1 = s[e]; e1 = e; }
    float w1  = __expf(l1 - l0);
    float inv = 1.f / (1.f + w1);
    int s0 = atomicAdd(&counts[e0], 1);
    lists[e0 * NTOK + s0] = tok;  gatew[e0 * NTOK + s0] = inv;
    int s1 = atomicAdd(&counts[e1], 1);
    lists[e1 * NTOK + s1] = tok;  gatew[e1 * NTOK + s1] = w1 * inv;
  }
}

// ---------------------------------------------------------------- K2: experts
__global__ void __launch_bounds__(256, 1)
moe_expert(const float* __restrict__ x, const float* __restrict__ Wfc,
           const float* __restrict__ Wproj, const int* __restrict__ counts,
           const int* __restrict__ lists, const float* __restrict__ gatew,
           float* __restrict__ out) {
  int e     = blockIdx.x;
  int count = counts[e];
  int tile0 = blockIdx.y * TM;
  if (tile0 >= count) return;                     // sparse: skip empty tiles

  __shared__ __align__(16) unsigned short xs[TM][128];     //  8 KB  x  (bf16)
  __shared__ __align__(16) unsigned short wfc_s[64][128];  // 16 KB  Wfc slice
  __shared__ __align__(16) unsigned short a_s[TM][64];     //  4 KB  activations
  __shared__ __align__(16) unsigned short wp_s[128][64];   // 16 KB  Wproj slice
  __shared__ int   tok_s[TM];
  __shared__ float g_s[TM];

  int tid  = threadIdx.x;
  int wave = tid >> 5, lane = tid & 31;
  if (tid < TM) {
    int idx = tile0 + tid;
    int t = 0; float g = 0.f;
    if (idx < count) { t = lists[e * NTOK + idx]; g = gatew[e * NTOK + idx]; }
    tok_s[tid] = t; g_s[tid] = g;                 // padded rows: gate 0 -> no-op
  }
  __syncthreads();

  const float* wfc_base = Wfc   + (size_t)e * HDIM * DDIM;   // [H][D]
  const float* wp_base  = Wproj + (size_t)e * DDIM * HDIM;   // [D][H]

  int mhalf  = wave & 1;        // 16-row half of the token tile
  int wgrp   = wave >> 1;       // 0..3
  int col    = lane & 15;
  int hi     = lane >> 4;       // lane half
  int kbaseA = hi ? 8 : 0;      // 16-bit A frag: K 0-7/16-23 vs 8-15/24-31
  int kbaseB = hi ? 8 : 0;      // 16-bit B frag: K 0-7/16-23 vs 8-15/24-31
  int rowl   = hi * 8;          // C/D frag: lanes>=16 hold rows M+8

  v8f zero = {};
  v8f acc[16];
#pragma unroll
  for (int i = 0; i < 16; ++i) acc[i] = zero;

  for (int hc = 0; hc < HDIM / 64; ++hc) {
    // ---------------- phase A: h[TM][64] = x * Wfc[hc-chunk]^T -------------
    v8f hacc = zero;
    for (int kb = 0; kb < DDIM; kb += 128) {
#pragma unroll
      for (int i = 0; i < 2; ++i) {               // stage x slice (b128)
        int g = i * 256 + tid;
        int r = g >> 4, c = (g & 15) << 3;
        *(u32x4*)&xs[r][c] = pack8(&x[(size_t)tok_s[r] * DDIM + kb + c]);
      }
#pragma unroll
      for (int i = 0; i < 4; ++i) {               // stage Wfc slice (b128)
        int g = i * 256 + tid;
        int r = g >> 4, c = (g & 15) << 3;
        const float* gp = &wfc_base[(size_t)(hc * 64 + r) * DDIM + kb + c];
        if (kb + 128 < DDIM) __builtin_prefetch(gp + 128, 0, 1);
        *(u32x4*)&wfc_s[r][c] = pack8(gp);
      }
      __syncthreads();
#pragma unroll
      for (int kk = 0; kk < 128; kk += 32) {
        Frag A, Bf;
        const unsigned short* ar = &xs[mhalf * 16 + col][0];
        A.q[0] = *(const u32x4*)(ar + kk + kbaseA);
        A.q[1] = *(const u32x4*)(ar + kk + kbaseA + 16);
        const unsigned short* br = &wfc_s[wgrp * 16 + col][0];
        Bf.q[0] = *(const u32x4*)(br + kk + kbaseB);
        Bf.q[1] = *(const u32x4*)(br + kk + kbaseB + 16);
        hacc = __builtin_amdgcn_wmma_f32_16x16x32_bf16(
                   false, A.v, false, Bf.v, (short)0, hacc, false, false);
      }
      __syncthreads();
    }
    // activation: sq(leaky_relu(h, 0.5)), pre-scaled by gate, -> bf16 LDS
#pragma unroll
    for (int r = 0; r < 8; ++r) {
      int row = mhalf * 16 + rowl + r;
      float h = hacc[r];
      float t = h >= 0.f ? h : 0.5f * h;
      float a = g_s[row] * t * t;
      a_s[row][wgrp * 16 + col] = (unsigned short)pack2(a, a);
    }
    __syncthreads();

    // phase-B A fragments for this chunk (K = 64 -> two k-steps)
    Frag A0, A1;
    {
      const unsigned short* ar = &a_s[mhalf * 16 + col][0];
      A0.q[0] = *(const u32x4*)(ar + kbaseA);
      A0.q[1] = *(const u32x4*)(ar + kbaseA + 16);
      A1.q[0] = *(const u32x4*)(ar + 32 + kbaseA);
      A1.q[1] = *(const u32x4*)(ar + 32 + kbaseA + 16);
    }
    // ---------------- phase B: out += a * Wproj[hc-chunk]^T ----------------
#pragma unroll
    for (int p = 0; p < 8; ++p) {                 // 128 d-cols each
#pragma unroll
      for (int i = 0; i < 4; ++i) {               // stage Wproj slice (b128)
        int g = i * 256 + tid;
        int r = g >> 3, c = (g & 7) << 3;
        *(u32x4*)&wp_s[r][c] =
            pack8(&wp_base[(size_t)(p * 128 + r) * HDIM + hc * 64 + c]);
      }
      __syncthreads();
#pragma unroll
      for (int j = 0; j < 2; ++j) {
        int nt = (wgrp << 1) | j;
        const unsigned short* br = &wp_s[nt * 16 + col][0];
        Frag B0, B1;
        B0.q[0] = *(const u32x4*)(br + kbaseB);
        B0.q[1] = *(const u32x4*)(br + kbaseB + 16);
        B1.q[0] = *(const u32x4*)(br + 32 + kbaseB);
        B1.q[1] = *(const u32x4*)(br + 32 + kbaseB + 16);
        v8f c0 = acc[(p << 1) | j];
        c0 = __builtin_amdgcn_wmma_f32_16x16x32_bf16(
                 false, A0.v, false, B0.v, (short)0, c0, false, false);
        c0 = __builtin_amdgcn_wmma_f32_16x16x32_bf16(
                 false, A1.v, false, B1.v, (short)0, c0, false, false);
        acc[(p << 1) | j] = c0;
      }
      __syncthreads();
    }
  }

  // -------- combine: f32 atomic add into scattered token rows --------------
#pragma unroll
  for (int p = 0; p < 8; ++p) {
#pragma unroll
    for (int j = 0; j < 2; ++j) {
      int cb = p * 128 + (((wgrp << 1) | j) * 16) + col;
      v8f c0 = acc[(p << 1) | j];
#pragma unroll
      for (int r = 0; r < 8; ++r) {
        int row = mhalf * 16 + rowl + r;
        atomicAdd(&out[(size_t)tok_s[row] * DDIM + cb], c0[r]);
      }
    }
  }
}

// ---------------------------------------------------------------- launcher
extern "C" void kernel_launch(void* const* d_in, const int* in_sizes, int n_in,
                              void* d_out, int out_size, void* d_ws, size_t ws_size,
                              hipStream_t stream) {
  const float* x     = (const float*)d_in[0];   // [B,T,D]
  const float* Wg    = (const float*)d_in[1];   // [E,D]
  const float* Wfc   = (const float*)d_in[2];   // [E,H,D]
  const float* Wproj = (const float*)d_in[3];   // [E,D,H]
  float* out = (float*)d_out;

  int*   counts = (int*)d_ws;
  int*   lists  = (int*)((char*)d_ws + 256);
  float* gatew  = (float*)((char*)d_ws + 256 + (size_t)NEXP * NTOK * 4);

  moe_zero<<<2048, 256, 0, stream>>>(out, out_size, counts);
  moe_gate<<<NTOK / 8, 256, 0, stream>>>(x, Wg, counts, lists, gatew);
  dim3 grid(NEXP, NTOK / TM);
  moe_expert<<<grid, 256, 0, stream>>>(x, Wfc, Wproj, counts, lists, gatew, out);
}
